// JaxAttentionWrapper_31963146617501
// MI455X (gfx1250) — matrix operand order repaired and checked by reading, not compiled, not run on hardware
//
#include <hip/hip_runtime.h>

// Problem constants (match reference).
#define BB   4
#define QLL  1024
#define LL   2048
#define HH   16
#define KVHH 4
#define DD   128
#define GG   4                      // HH / KVHH
#define SCALE 0.08838834764831845f  // 1/sqrt(128)
#define NEGI (-1e30f)

typedef __bf16 bf16;
typedef __attribute__((ext_vector_type(16))) __bf16 v16bf;
typedef __attribute__((ext_vector_type(8)))  __bf16 v8bf;
typedef __attribute__((ext_vector_type(8)))  float  v8f;
typedef __attribute__((ext_vector_type(4)))  int    v4i;

// ---- async global->LDS path (CDNA5), guarded so compile never breaks ------
#if defined(__gfx1250__) && __has_builtin(__builtin_amdgcn_global_load_async_to_lds_b128)
#define HAVE_ASYNC_LDS 1
#else
#define HAVE_ASYNC_LDS 0
#endif

#if HAVE_ASYNC_LDS
__device__ __forceinline__ void async_b128(const void* g, void* l) {
  // Builtin takes generic pointers to 16-byte vectors (per hipcc diagnostic):
  // (v4i* global_src, v4i* lds_dst, imm offset, imm cpol)
  __builtin_amdgcn_global_load_async_to_lds_b128((v4i*)g, (v4i*)l, 0, 0);
}
__device__ __forceinline__ void wait_async0() {
#if __has_builtin(__builtin_amdgcn_s_wait_asynccnt)
  __builtin_amdgcn_s_wait_asynccnt(0);
#else
  asm volatile("s_wait_asynccnt 0x0" ::: "memory");
#endif
}
#endif

// ---------------------------------------------------------------------------
// Kernel 1: merge new k/v tokens into the cache copies (ragged scatter).
// Optionally also emits bf16 K ([key][d]) and transposed bf16 V^T ([d][key])
// caches into workspace for the WMMA attention kernel.
// ---------------------------------------------------------------------------
template <bool WRITE_BF16>
__global__ __launch_bounds__(256) void cache_merge_kernel(
    const float* __restrict__ knew, const float* __restrict__ vnew,
    const float* __restrict__ cache_k, const float* __restrict__ cache_v,
    const int* __restrict__ seq_lens,
    float* __restrict__ new_ck, float* __restrict__ new_cv,
    bf16* __restrict__ wsK, bf16* __restrict__ wsVt) {
  const long total4 = (long)BB * KVHH * LL * DD / 4;
  long idx4 = (long)blockIdx.x * blockDim.x + threadIdx.x;
  if (idx4 >= total4) return;
  const long e = idx4 * 4;
  const int d = (int)(e % DD);           // multiple of 4
  long r = e / DD;
  const int l = (int)(r % LL);
  r /= LL;
  const int kvh = (int)(r % KVHH);
  const int b = (int)(r / KVHH);
  const int s = seq_lens[b];

  float4 kq, vq;
  if (l >= s && l < s + QLL) {
    const int t = l - s;
    const long src = ((long)(b * QLL + t)) * (KVHH * DD) + kvh * DD + d;
    kq = *(const float4*)(knew + src);
    vq = *(const float4*)(vnew + src);
  } else {
    kq = *(const float4*)(cache_k + e);
    vq = *(const float4*)(cache_v + e);
  }
  *(float4*)(new_ck + e) = kq;
  *(float4*)(new_cv + e) = vq;

  if (WRITE_BF16) {
    wsK[e + 0] = (bf16)kq.x;
    wsK[e + 1] = (bf16)kq.y;
    wsK[e + 2] = (bf16)kq.z;
    wsK[e + 3] = (bf16)kq.w;
    const long vtb = ((long)(b * KVHH + kvh) * DD + d) * LL + l;
    wsVt[vtb + 0 * LL] = (bf16)vq.x;
    wsVt[vtb + 1 * LL] = (bf16)vq.y;
    wsVt[vtb + 2 * LL] = (bf16)vq.z;
    wsVt[vtb + 3 * LL] = (bf16)vq.w;
  }
}

// ---------------------------------------------------------------------------
// Online-softmax update over a 16x32 score tile in WMMA C layout.
// MASKED=false is the hot path (wave-uniform branch) with no cndmask work.
// ---------------------------------------------------------------------------
template <bool MASKED>
__device__ __forceinline__ void softmax_update(
    v8f& s0, v8f& s1, float* __restrict__ mrow, float* __restrict__ srow,
    v8f (&o)[8], bf16* __restrict__ pP, int hl, int ln, int limitBase,
    int kbase) {
#pragma unroll
  for (int r = 0; r < 8; ++r) {
    float a0 = s0[r], a1 = s1[r];
    if (MASKED) {
      const int limit = limitBase + r + 8 * hl;  // key valid iff key <= limit
      a0 = (kbase + ln <= limit) ? a0 : NEGI;
      a1 = (kbase + 16 + ln <= limit) ? a1 : NEGI;
    }
    float mx = fmaxf(a0, a1);
    mx = fmaxf(mx, __shfl_xor(mx, 1, 32));
    mx = fmaxf(mx, __shfl_xor(mx, 2, 32));
    mx = fmaxf(mx, __shfl_xor(mx, 4, 32));
    mx = fmaxf(mx, __shfl_xor(mx, 8, 32));
    const float mn = fmaxf(mrow[r], mx);
    const float sc = __expf(mrow[r] - mn);
    float p0, p1;
    if (MASKED) {
      p0 = (a0 <= -1e29f) ? 0.f : __expf(a0 - mn);
      p1 = (a1 <= -1e29f) ? 0.f : __expf(a1 - mn);
    } else {
      p0 = __expf(a0 - mn);
      p1 = __expf(a1 - mn);
    }
    float ps = p0 + p1;
    ps += __shfl_xor(ps, 1, 32);
    ps += __shfl_xor(ps, 2, 32);
    ps += __shfl_xor(ps, 4, 32);
    ps += __shfl_xor(ps, 8, 32);
    srow[r] = srow[r] * sc + ps;
    mrow[r] = mn;
    const int row = r + 8 * hl;
    pP[row * 32 + ln] = (bf16)p0;
    pP[row * 32 + 16 + ln] = (bf16)p1;
#pragma unroll
    for (int n = 0; n < 8; ++n) o[n][r] *= sc;
  }
}

// ---------------------------------------------------------------------------
// Kernel 2 (primary): flash attention, bf16 WMMA, async-to-LDS staging from
// the pre-converted bf16 K / V^T workspace caches.
// Block = 256 threads = 8 wave32s -> (b, kvh, g, 128-query chunk).
// ---------------------------------------------------------------------------
__global__ __launch_bounds__(256) void attn_kernel_bf16(
    const float* __restrict__ q, const int* __restrict__ seq_lens,
    const bf16* __restrict__ wsK, const bf16* __restrict__ wsVt,
    float* __restrict__ out) {
  __shared__ __attribute__((aligned(32))) bf16 ldsK[32 * 128];   // [key][d]
  __shared__ __attribute__((aligned(32))) bf16 ldsVt[128 * 32];  // [d][key]
  __shared__ __attribute__((aligned(32))) bf16 ldsP[8 * 16 * 32];

  const int tid  = threadIdx.x;
  const int wv   = tid >> 5;
  const int lane = tid & 31;
  const int hl   = lane >> 4;
  const int ln   = lane & 15;

  const int bid  = blockIdx.x;
  const int chnk = bid & 7;
  const int g    = (bid >> 3) & (GG - 1);
  const int kvh  = (bid >> 5) & (KVHH - 1);
  const int b    = bid >> 7;

  const int h  = kvh * GG + g;
  const int t0 = chnk * 128 + wv * 16;
  const int s  = seq_lens[b];

  // Q tile 16x128 -> 4 bf16 A-fragments (16x32), pre-scaled.
  const float* qrow = q + ((long)(b * QLL + t0 + ln)) * (HH * DD) + h * DD;
  v16bf qa[4];
#pragma unroll
  for (int c = 0; c < 4; ++c) {
#pragma unroll
    for (int e = 0; e < 8; ++e) {
      qa[c][e]     = (bf16)(qrow[c * 32 + hl * 16 + e] * SCALE);
      qa[c][8 + e] = (bf16)(qrow[c * 32 + hl * 16 + 8 + e] * SCALE);
    }
  }

  v8f o[8];
#pragma unroll
  for (int n = 0; n < 8; ++n) o[n] = (v8f){0.f,0.f,0.f,0.f,0.f,0.f,0.f,0.f};
  float mrow[8], srow[8];
#pragma unroll
  for (int r = 0; r < 8; ++r) { mrow[r] = NEGI; srow[r] = 0.f; }

  const bf16* KgB  = wsK  + ((long)(b * KVHH + kvh)) * LL * DD;  // [key][d]
  const bf16* VtgB = wsVt + ((long)(b * KVHH + kvh)) * DD * LL;  // [d][key]
  const int kmax = s + chnk * 128 + 127;
  const int nblk = (kmax >> 5) + 1;

  // Per-thread staging assignment (same for async and fallback paths):
  //   K : 8KB contiguous tile; thread covers 16B at tid*8 elems (+2048 elems)
  //   V^T: row d = tid>>1, 16 contiguous keys at (tid&1)*16
  const int vd   = tid >> 1;
  const int voff = (tid & 1) * 16;
  bf16* ldsVrow = &ldsVt[vd * 32 + voff];
  bf16* pP = &ldsP[wv * 16 * 32];

  for (int blk = 0; blk < nblk; ++blk) {
    __syncthreads();
    const bf16* kg = KgB + (long)blk * 32 * DD;
    const bf16* vg = VtgB + (long)vd * LL + blk * 32 + voff;
#if HAVE_ASYNC_LDS
    async_b128(kg + tid * 8, &ldsK[tid * 8]);
    async_b128(kg + 2048 + tid * 8, &ldsK[2048 + tid * 8]);
    async_b128(vg, ldsVrow);
    async_b128(vg + 8, ldsVrow + 8);
    wait_async0();
#else
    {
      const int4 ka = *(const int4*)(kg + tid * 8);
      const int4 kb2 = *(const int4*)(kg + 2048 + tid * 8);
      *(int4*)&ldsK[tid * 8] = ka;
      *(int4*)&ldsK[2048 + tid * 8] = kb2;
      const int4 va = *(const int4*)(vg);
      const int4 vb2 = *(const int4*)(vg + 8);
      *(int4*)ldsVrow = va;
      *(int4*)(ldsVrow + 8) = vb2;
    }
#endif
    __syncthreads();

    if (blk + 1 < nblk) {  // hint next key block toward L2
      __builtin_prefetch(KgB + (long)(blk + 1) * 32 * DD + tid * 32, 0, 1);
    }

    // ---- S = Q*K^T : 16x32 scores, 8 WMMAs -----------------------------
    v8f s0 = (v8f){0.f,0.f,0.f,0.f,0.f,0.f,0.f,0.f};
    v8f s1 = (v8f){0.f,0.f,0.f,0.f,0.f,0.f,0.f,0.f};
#pragma unroll
    for (int c = 0; c < 4; ++c) {
      const v16bf kb0 = *(const v16bf*)&ldsK[ln * 128 + c * 32 + hl * 16];
      const v16bf kb1 = *(const v16bf*)&ldsK[(16 + ln) * 128 + c * 32 + hl * 16];
      s0 = __builtin_amdgcn_wmma_f32_16x16x32_bf16(false, qa[c], false, kb0,
                                                   (short)0, s0, false, false);
      s1 = __builtin_amdgcn_wmma_f32_16x16x32_bf16(false, qa[c], false, kb1,
                                                   (short)0, s1, false, false);
    }

    // ---- online softmax: unmasked fast path for interior blocks --------
    if (blk * 32 + 31 <= s + t0) {
      softmax_update<false>(s0, s1, mrow, srow, o, pP, hl, ln, s + t0,
                            blk * 32);
    } else {
      softmax_update<true>(s0, s1, mrow, srow, o, pP, hl, ln, s + t0,
                           blk * 32);
    }

    // ---- P re-layout (C -> A frag) via per-wave LDS bounce -------------
    v16bf pa;
    {
      const v8bf plo = *(const v8bf*)&pP[ln * 32 + hl * 16];
      const v8bf phi = *(const v8bf*)&pP[ln * 32 + hl * 16 + 8];
#pragma unroll
      for (int e = 0; e < 8; ++e) { pa[e] = plo[e]; pa[8 + e] = phi[e]; }
    }

    // ---- O += P*V : 8 WMMAs --------------------------------------------
#pragma unroll
    for (int n = 0; n < 8; ++n) {
      const v16bf vb = *(const v16bf*)&ldsVt[(n * 16 + ln) * 32 + hl * 16];
      o[n] = __builtin_amdgcn_wmma_f32_16x16x32_bf16(false, pa, false, vb,
                                                     (short)0, o[n], false, false);
    }
  }

#pragma unroll
  for (int r = 0; r < 8; ++r) {
    const int t = t0 + r + 8 * hl;
    const float inv = 1.0f / srow[r];
    const long obase = ((long)(b * QLL + t)) * (HH * DD) + h * DD;
#pragma unroll
    for (int n = 0; n < 8; ++n) {
      out[obase + n * 16 + ln] = o[n][r] * inv;
    }
  }
}

// ---------------------------------------------------------------------------
// Kernel 2 (fallback, used only if workspace is too small): identical math,
// but stages from the f32 merged caches with in-kernel cvt + V transpose.
// ---------------------------------------------------------------------------
__global__ __launch_bounds__(256) void attn_kernel_f32(
    const float* __restrict__ q, const int* __restrict__ seq_lens,
    const float* __restrict__ ck, const float* __restrict__ cv,
    float* __restrict__ out) {
  __shared__ __attribute__((aligned(32))) bf16 ldsK[32 * 128];
  __shared__ __attribute__((aligned(32))) bf16 ldsVt[128 * 32];
  __shared__ __attribute__((aligned(32))) bf16 ldsP[8 * 16 * 32];

  const int tid  = threadIdx.x;
  const int wv   = tid >> 5;
  const int lane = tid & 31;
  const int hl   = lane >> 4;
  const int ln   = lane & 15;

  const int bid  = blockIdx.x;
  const int chnk = bid & 7;
  const int g    = (bid >> 3) & (GG - 1);
  const int kvh  = (bid >> 5) & (KVHH - 1);
  const int b    = bid >> 7;

  const int h  = kvh * GG + g;
  const int t0 = chnk * 128 + wv * 16;
  const int s  = seq_lens[b];

  const float* qrow = q + ((long)(b * QLL + t0 + ln)) * (HH * DD) + h * DD;
  v16bf qa[4];
#pragma unroll
  for (int c = 0; c < 4; ++c) {
#pragma unroll
    for (int e = 0; e < 8; ++e) {
      qa[c][e]     = (bf16)(qrow[c * 32 + hl * 16 + e] * SCALE);
      qa[c][8 + e] = (bf16)(qrow[c * 32 + hl * 16 + 8 + e] * SCALE);
    }
  }

  v8f o[8];
#pragma unroll
  for (int n = 0; n < 8; ++n) o[n] = (v8f){0.f,0.f,0.f,0.f,0.f,0.f,0.f,0.f};
  float mrow[8], srow[8];
#pragma unroll
  for (int r = 0; r < 8; ++r) { mrow[r] = NEGI; srow[r] = 0.f; }

  const long cbase = ((long)(b * KVHH + kvh)) * LL * DD;
  const float* Kg = ck + cbase;
  const float* Vg = cv + cbase;
  const int kmax = s + chnk * 128 + 127;
  const int nblk = (kmax >> 5) + 1;
  bf16* pP = &ldsP[wv * 16 * 32];

  for (int blk = 0; blk < nblk; ++blk) {
    __syncthreads();
#pragma unroll
    for (int rep = 0; rep < 4; ++rep) {
      const int f   = tid + rep * 256;
      const int row = f >> 5;
      const int dd  = (f & 31) * 4;
      const long ga = (long)(blk * 32 + row) * DD + dd;
      const float4 kq = *(const float4*)(Kg + ga);
      const float4 vq = *(const float4*)(Vg + ga);
      ldsK[row * 128 + dd + 0] = (bf16)kq.x;
      ldsK[row * 128 + dd + 1] = (bf16)kq.y;
      ldsK[row * 128 + dd + 2] = (bf16)kq.z;
      ldsK[row * 128 + dd + 3] = (bf16)kq.w;
      ldsVt[(dd + 0) * 32 + row] = (bf16)vq.x;
      ldsVt[(dd + 1) * 32 + row] = (bf16)vq.y;
      ldsVt[(dd + 2) * 32 + row] = (bf16)vq.z;
      ldsVt[(dd + 3) * 32 + row] = (bf16)vq.w;
    }
    __syncthreads();

    v8f s0 = (v8f){0.f,0.f,0.f,0.f,0.f,0.f,0.f,0.f};
    v8f s1 = (v8f){0.f,0.f,0.f,0.f,0.f,0.f,0.f,0.f};
#pragma unroll
    for (int c = 0; c < 4; ++c) {
      const v16bf kb0 = *(const v16bf*)&ldsK[ln * 128 + c * 32 + hl * 16];
      const v16bf kb1 = *(const v16bf*)&ldsK[(16 + ln) * 128 + c * 32 + hl * 16];
      s0 = __builtin_amdgcn_wmma_f32_16x16x32_bf16(false, qa[c], false, kb0,
                                                   (short)0, s0, false, false);
      s1 = __builtin_amdgcn_wmma_f32_16x16x32_bf16(false, qa[c], false, kb1,
                                                   (short)0, s1, false, false);
    }

    if (blk * 32 + 31 <= s + t0) {
      softmax_update<false>(s0, s1, mrow, srow, o, pP, hl, ln, s + t0,
                            blk * 32);
    } else {
      softmax_update<true>(s0, s1, mrow, srow, o, pP, hl, ln, s + t0,
                           blk * 32);
    }

    v16bf pa;
    {
      const v8bf plo = *(const v8bf*)&pP[ln * 32 + hl * 16];
      const v8bf phi = *(const v8bf*)&pP[ln * 32 + hl * 16 + 8];
#pragma unroll
      for (int e = 0; e < 8; ++e) { pa[e] = plo[e]; pa[8 + e] = phi[e]; }
    }
#pragma unroll
    for (int n = 0; n < 8; ++n) {
      const v16bf vb = *(const v16bf*)&ldsVt[(n * 16 + ln) * 32 + hl * 16];
      o[n] = __builtin_amdgcn_wmma_f32_16x16x32_bf16(false, pa, false, vb,
                                                     (short)0, o[n], false, false);
    }
  }

#pragma unroll
  for (int r = 0; r < 8; ++r) {
    const int t = t0 + r + 8 * hl;
    const float inv = 1.0f / srow[r];
    const long obase = ((long)(b * QLL + t)) * (HH * DD) + h * DD;
#pragma unroll
    for (int n = 0; n < 8; ++n) {
      out[obase + n * 16 + ln] = o[n][r] * inv;
    }
  }
}

// ---------------------------------------------------------------------------
extern "C" void kernel_launch(void* const* d_in, const int* in_sizes, int n_in,
                              void* d_out, int out_size, void* d_ws, size_t ws_size,
                              hipStream_t stream) {
  const float* q       = (const float*)d_in[0];
  const float* k       = (const float*)d_in[1];
  const float* v       = (const float*)d_in[2];
  const float* cache_k = (const float*)d_in[3];
  const float* cache_v = (const float*)d_in[4];
  const int*   seqlens = (const int*)d_in[5];

  float* out    = (float*)d_out;
  float* new_ck = out + (long)BB * QLL * HH * DD;
  float* new_cv = new_ck + (long)BB * KVHH * LL * DD;

  const long cacheElems = (long)BB * KVHH * LL * DD;        // 4M
  const size_t wsNeeded = 2ull * cacheElems * sizeof(bf16); // 16 MB
  bf16* wsK  = (bf16*)d_ws;
  bf16* wsVt = wsK + cacheElems;

  const long total4 = cacheElems / 4;
  const int mergeGrid = (int)((total4 + 255) / 256);
  const int attnGrid = BB * KVHH * GG * (QLL / 128);  // 512 blocks

  if (ws_size >= wsNeeded) {
    cache_merge_kernel<true><<<mergeGrid, 256, 0, stream>>>(
        k, v, cache_k, cache_v, seqlens, new_ck, new_cv, wsK, wsVt);
    attn_kernel_bf16<<<attnGrid, 256, 0, stream>>>(q, seqlens, wsK, wsVt, out);
  } else {
    cache_merge_kernel<false><<<mergeGrid, 256, 0, stream>>>(
        k, v, cache_k, cache_v, seqlens, new_ck, new_cv, nullptr, nullptr);
    attn_kernel_f32<<<attnGrid, 256, 0, stream>>>(q, seqlens, new_ck, new_cv, out);
  }
}
